// AbsDiagNet_10599979287001
// MI455X (gfx1250) — compile-verified, matrix operand order checked by reading, not run on hardware
//
#include <hip/hip_runtime.h>

#define S_LEN 2048
#define B_DIM 128
#define I_DIM 256
#define H_DIM 1024
#define O_DIM 256

typedef __attribute__((ext_vector_type(16))) __bf16 v16bf;
typedef __attribute__((ext_vector_type(8)))  __bf16 v8bf;
typedef __attribute__((ext_vector_type(8)))  float  v8f;

// ---- CDNA5 async global->LDS path (guarded; falls back to load+ds_store) ----
#if defined(__HIP_DEVICE_COMPILE__) && defined(__gfx1250__) && \
    __has_builtin(__builtin_amdgcn_global_load_async_to_lds_b128)
#define USE_ASYNC_LDS 1
#endif

#if __has_builtin(__builtin_amdgcn_s_wait_asynccnt)
#define WAIT_ASYNCCNT(n) __builtin_amdgcn_s_wait_asynccnt(n)
#else
#define WAIT_ASYNCCNT(n) asm volatile("s_wait_asynccnt " #n ::: "memory")
#endif

#ifdef USE_ASYNC_LDS
// Builtin signature (from hipcc diagnostic): param0 is
//   'int __attribute__((vector_size(16))) __device__ *'  (global int4*)
// so pass exactly-typed AS(1)/AS(3) int4 pointers.
typedef int v4i_g __attribute__((vector_size(16)));
#define ASG __attribute__((address_space(1)))
#define ASL __attribute__((address_space(3)))

__device__ __forceinline__ void async_g2l_16B(const float* g, float* l) {
  __builtin_amdgcn_global_load_async_to_lds_b128(
      (ASG v4i_g*)g,
      (ASL v4i_g*)l,
      0, 0);
}
#endif

__device__ __forceinline__ v16bf cvt16_bf(const float* t) {
  v16bf r;
#pragma unroll
  for (int i = 0; i < 16; ++i) r[i] = (__bf16)t[i];
  return r;
}

__device__ __forceinline__ v16bf join8(v8bf lo, v8bf hi) {
  v16bf a;
#pragma unroll
  for (int i = 0; i < 8; ++i) { a[i] = lo[i]; a[i + 8] = hi[i]; }
  return a;
}

// ---------------------------------------------------------------------------
// Kernel 1: fused  pre = X @ W_ih^T  +  scan  h = |pre + hh*h|
// Grid: 16 blocks (8 batch-tiles x 2 h-groups), 512 threads (16 waves).
// Each wave owns two 16x16 (batch x h) accumulator tiles held in VGPRs
// across all 2048 timesteps. W_ih fragments (bf16 B-matrix layout) are
// register-resident. X tiles stream through LDS (async double buffer).
// ---------------------------------------------------------------------------
__global__ __launch_bounds__(512)
void absdiag_scan_kernel(const float* __restrict__ X,
                         const float* __restrict__ W_ih,
                         const float* __restrict__ hh,
                         __bf16* __restrict__ hout)
{
#ifdef USE_ASYNC_LDS
  __shared__ float xstage[2][16][256];   // 32 KB double-buffered f32 staging
#endif
  __shared__ __bf16 xbf[16][264];        // bf16 A-tile, padded row stride

  const int tid   = threadIdx.x;
  const int lane  = tid & 31;
  const int wave  = tid >> 5;
  const int lmod  = lane & 15;   // N (B-frag) / M (A-frag) row within tile
  const int lhalf = lane >> 4;   // K-half selector
  const int bt = blockIdx.x >> 1;       // batch tile 0..7
  const int hg = blockIdx.x & 1;        // h group 0..1 (512 h each)

  const int hbase0 = (hg * 32 + wave * 2) * 16;
  const int hbase1 = hbase0 + 16;

  // per-thread slice of the 16x256 f32 X tile (32 bytes)
  const int crow = tid >> 5;            // 0..15
  const int ci0  = (tid & 31) * 8;      // 0..248
  const float* Xrow = X + ((size_t)bt * 16 + crow) * I_DIM + ci0;

  // ---- preload W_ih fragments (B-matrix layout: lane=N=h row, 16 contiguous K) ----
  v16bf w0[8], w1[8];
#pragma unroll
  for (int k = 0; k < 8; ++k) {
    const float* p0 = W_ih + (size_t)(hbase0 + lmod) * I_DIM + 32 * k + 16 * lhalf;
    const float* p1 = W_ih + (size_t)(hbase1 + lmod) * I_DIM + 32 * k + 16 * lhalf;
    float t0[16], t1[16];
#pragma unroll
    for (int q = 0; q < 4; ++q) {
      *(float4*)&t0[4 * q] = ((const float4*)p0)[q];
      *(float4*)&t1[4 * q] = ((const float4*)p1)[q];
    }
    w0[k] = cvt16_bf(t0);
    w1[k] = cvt16_bf(t1);
  }

  const float hh0 = hh[hbase0 + lmod];
  const float hh1 = hh[hbase1 + lmod];

  v8f h0 = {};   // recurrent state, C/D layout, zero-initialized
  v8f h1 = {};

#ifdef USE_ASYNC_LDS
#pragma unroll
  for (int p = 0; p < 2; ++p) {        // prefetch steps 0,1
    const float* g = Xrow + (size_t)p * B_DIM * I_DIM;
    async_g2l_16B(g,     &xstage[p][crow][ci0]);
    async_g2l_16B(g + 4, &xstage[p][crow][ci0 + 4]);
  }
#endif

  for (int s = 0; s < S_LEN; ++s) {
    float t[8];
#ifdef USE_ASYNC_LDS
    if (s + 1 < S_LEN) { WAIT_ASYNCCNT(2); } else { WAIT_ASYNCCNT(0); }
    __syncthreads();                   // step s-1 consumers of xbf are done
    *(float4*)&t[0] = *(const float4*)&xstage[s & 1][crow][ci0];
    *(float4*)&t[4] = *(const float4*)&xstage[s & 1][crow][ci0 + 4];
#else
    __syncthreads();
    const float* gsrc = Xrow + (size_t)s * B_DIM * I_DIM;
    *(float4*)&t[0] = ((const float4*)gsrc)[0];
    *(float4*)&t[4] = ((const float4*)gsrc)[1];
#endif
    v8bf vb;
#pragma unroll
    for (int j = 0; j < 8; ++j) vb[j] = (__bf16)t[j];
    *(v8bf*)&xbf[crow][ci0] = vb;      // one ds_store_b128
    __syncthreads();
#ifdef USE_ASYNC_LDS
    if (s + 2 < S_LEN) {               // refill the buffer just consumed
      const float* g = Xrow + (size_t)(s + 2) * B_DIM * I_DIM;
      async_g2l_16B(g,     &xstage[s & 1][crow][ci0]);
      async_g2l_16B(g + 4, &xstage[s & 1][crow][ci0 + 4]);
    }
#endif

    // C = hh * h ; D = sum_k A_k B_k + C ; h = |D|
    v8f c0, c1;
#pragma unroll
    for (int r = 0; r < 8; ++r) { c0[r] = hh0 * h0[r]; c1[r] = hh1 * h1[r]; }
#pragma unroll
    for (int k = 0; k < 8; ++k) {
      const __bf16* pa = &xbf[lmod][32 * k + 16 * lhalf];
      v16bf a = join8(*(const v8bf*)pa, *(const v8bf*)(pa + 8));
      c0 = __builtin_amdgcn_wmma_f32_16x16x32_bf16(false, a, false, w0[k],
                                                   (short)0, c0, false, false);
      c1 = __builtin_amdgcn_wmma_f32_16x16x32_bf16(false, a, false, w1[k],
                                                   (short)0, c1, false, false);
    }
#pragma unroll
    for (int r = 0; r < 8; ++r) { h0[r] = fabsf(c0[r]); h1[r] = fabsf(c1[r]); }
  }

  // store h_final as bf16 row-major [B][H] into workspace
#pragma unroll
  for (int r = 0; r < 8; ++r) {
    const int b = bt * 16 + r + 8 * lhalf;   // C/D layout: M = r + 8*(lane>=16)
    hout[(size_t)b * H_DIM + hbase0 + lmod] = (__bf16)h0[r];
    hout[(size_t)b * H_DIM + hbase1 + lmod] = (__bf16)h1[r];
  }
}

// ---------------------------------------------------------------------------
// Kernel 2: out = h_final @ W_ho^T + b_ho   ([128,1024]x[1024,256] -> [128,256])
// 128 single-wave blocks, one 16x16 output tile each, K=1024 chained WMMA.
// ---------------------------------------------------------------------------
__global__ __launch_bounds__(32)
void out_proj_kernel(const __bf16* __restrict__ hbf,
                     const float* __restrict__ W_ho,
                     const float* __restrict__ b_ho,
                     float* __restrict__ out)
{
  const int lane  = threadIdx.x & 31;
  const int lmod  = lane & 15;
  const int lhalf = lane >> 4;
  const int ot = blockIdx.x & 15;   // 16 output tiles
  const int bt = blockIdx.x >> 4;   // 8 batch tiles
  const int obase = ot * 16;

  v8f c = {};
#pragma unroll 4
  for (int k = 0; k < H_DIM / 32; ++k) {
    const __bf16* pa = hbf + (size_t)(bt * 16 + lmod) * H_DIM + 32 * k + 16 * lhalf;
    v16bf a = join8(*(const v8bf*)pa, *(const v8bf*)(pa + 8));
    const float* pb = W_ho + (size_t)(obase + lmod) * H_DIM + 32 * k + 16 * lhalf;
    float tb[16];
#pragma unroll
    for (int q = 0; q < 4; ++q) *(float4*)&tb[4 * q] = ((const float4*)pb)[q];
    v16bf b = cvt16_bf(tb);
    c = __builtin_amdgcn_wmma_f32_16x16x32_bf16(false, a, false, b,
                                                (short)0, c, false, false);
  }
  const float bias = b_ho[obase + lmod];
#pragma unroll
  for (int r = 0; r < 8; ++r) {
    const int b = bt * 16 + r + 8 * lhalf;
    out[(size_t)b * O_DIM + obase + lmod] = c[r] + bias;
  }
}

// ---------------------------------------------------------------------------
extern "C" void kernel_launch(void* const* d_in, const int* in_sizes, int n_in,
                              void* d_out, int out_size, void* d_ws, size_t ws_size,
                              hipStream_t stream) {
  (void)in_sizes; (void)n_in; (void)out_size; (void)ws_size;
  const float* X    = (const float*)d_in[0];   // [2048,128,256]
  const float* W_ih = (const float*)d_in[1];   // [1024,256]
  const float* hh   = (const float*)d_in[2];   // [1024]
  const float* W_ho = (const float*)d_in[3];   // [256,1024]
  const float* b_ho = (const float*)d_in[4];   // [256]
  float* out = (float*)d_out;                  // [128,256]
  __bf16* hbf = (__bf16*)d_ws;                 // 128*1024 bf16 = 256 KB scratch

  absdiag_scan_kernel<<<dim3(16), dim3(512), 0, stream>>>(X, W_ih, hh, hbf);
  out_proj_kernel<<<dim3(128), dim3(32), 0, stream>>>(hbf, W_ho, b_ho, out);
}
